// CrissCrossAttention_15161234555171
// MI455X (gfx1250) — matrix operand order repaired and checked by reading, not compiled, not run on hardware
//
#include <hip/hip_runtime.h>
#include <hip/hip_bf16.h>

typedef __bf16 bf16;
typedef __attribute__((ext_vector_type(16))) __bf16 v16bf;
typedef __attribute__((ext_vector_type(8)))  float  v8f;

union FragU { v16bf v; uint4 q[2]; };

// ---------------------------------------------------------------------------
// CDNA5 async copy: global -> LDS, 16B per lane, tracked by ASYNCcnt.
// LDS operand is the 32-bit LDS address (low 32 bits of the generic pointer).
// ---------------------------------------------------------------------------
__device__ __forceinline__ void async_ld_b128(void* lds, const void* gsrc) {
  asm volatile("global_load_async_to_lds_b128 %0, %1, off"
               :: "v"((unsigned)(unsigned long long)lds),
                  "v"((unsigned long long)gsrc)
               : "memory");
}
__device__ __forceinline__ void async_st_b128(void* gdst, const void* lds) {
  asm volatile("global_store_async_from_lds_b128 %0, %1, off"
               :: "v"((unsigned long long)gdst),
                  "v"((unsigned)(unsigned long long)lds)
               : "memory");
}
__device__ __forceinline__ void wait_async0() {
  asm volatile("s_wait_asynccnt 0x0" ::: "memory");
}

// A fragment: matrix stored row-major [m][k], k contiguous.
__device__ __forceinline__ v16bf frag_a(const bf16* base, int ldk) {
  int lane = threadIdx.x & 31;
  int g = lane >> 4, m = lane & 15;
  const bf16* p = base + m * ldk + g * 8;
  FragU f;
  f.q[0] = *(const uint4*)(p);
  f.q[1] = *(const uint4*)(p + 16);
  return f.v;
}

// B fragment: matrix stored as [n][k], k contiguous (B^T row-major).
__device__ __forceinline__ v16bf frag_b(const bf16* base, int ldk) {
  int lane = threadIdx.x & 31;
  int g = lane >> 4, n = lane & 15;
  const bf16* p = base + n * ldk + g * 8;
  FragU f;
  f.q[0] = *(const uint4*)(p);
  f.q[1] = *(const uint4*)(p + 16);
  return f.v;
}

__device__ __forceinline__ v8f wmma_bf16(v16bf a, v16bf b, v8f c) {
  return __builtin_amdgcn_wmma_f32_16x16x32_bf16(false, a, false, b, (short)0, c,
                                                 false, false);
}

// ---------------------------------------------------------------------------
#define BB 16
#define CC 512
#define HH 96
#define WW 96
#define NP (HH * WW)
#define CQ 64
#define OC 640

// ---------------------------------------------------------------------------
// K0a: pack Wq|Wk|Wv -> bf16 [640][512], biases -> f32 [640]
// ---------------------------------------------------------------------------
__global__ __launch_bounds__(256) void pack_weights_kernel(
    const float* __restrict__ Wq, const float* __restrict__ bq,
    const float* __restrict__ Wk, const float* __restrict__ bk,
    const float* __restrict__ Wv, const float* __restrict__ bv,
    bf16* __restrict__ Wqkv, float* __restrict__ bqkv) {
  int idx = blockIdx.x * 256 + threadIdx.x;
  if (idx < OC * CC) {
    int o = idx >> 9, c = idx & 511;
    float v;
    if (o < 64)       v = Wq[o * CC + c];
    else if (o < 128) v = Wk[(o - 64) * CC + c];
    else              v = Wv[(o - 128) * CC + c];
    Wqkv[idx] = (bf16)v;
  }
  if (idx < OC) {
    float v = (idx < 64) ? bq[idx] : (idx < 128) ? bk[idx - 64] : bv[idx - 128];
    bqkv[idx] = v;
  }
}

// ---------------------------------------------------------------------------
// K0b: x [b][c][p] f32 -> Xt [b][p][c] bf16  (async-load 32x32 tile, transpose)
// ---------------------------------------------------------------------------
__global__ __launch_bounds__(256) void transpose_x_kernel(
    const float* __restrict__ x, bf16* __restrict__ Xt) {
  __shared__ __align__(16) float tile[32 * 36];
  const int tid = threadIdx.x;
  const int p0 = blockIdx.x * 32, c0 = blockIdx.y * 32, b = blockIdx.z;
  {
    int row = tid >> 3, seg = tid & 7;  // 32 rows x 8 segs of 4 floats
    async_ld_b128(tile + row * 36 + seg * 4,
                  x + ((size_t)b * CC + c0 + row) * NP + p0 + seg * 4);
  }
  wait_async0();
  __syncthreads();
#pragma unroll
  for (int i = 0; i < 2; ++i) {
    int e = tid + i * 256;
    int pr = e >> 4, cp = (e & 15) * 2;
    union { bf16 h[2]; unsigned u; } pk;
    pk.h[0] = (bf16)tile[cp * 36 + pr];
    pk.h[1] = (bf16)tile[(cp + 1) * 36 + pr];
    *(unsigned*)(Xt + ((size_t)b * NP + p0 + pr) * CC + c0 + cp) = pk.u;
  }
}

// ---------------------------------------------------------------------------
// K1: QKV GEMM. Out[o,p] = sum_c Wqkv[o][c]*Xt[p][c] + bias[o]
// Block tile 128(M) x 64(N), 8 waves in 4x2 grid, 2x2 16x16 tiles per wave.
// Double-buffered async-to-LDS staging of A (128x32) and B (64x32) per k-step.
//   y==0  -> q + k rows, stored [b][p][64] c-contiguous
//   y>=1  -> v rows, LDS-transposed then async-stored from LDS to vh[b][c][h][w]
// grid: (NP/64, OC/128, B)
// ---------------------------------------------------------------------------
__global__ __launch_bounds__(256) void qkv_gemm_kernel(
    const bf16* __restrict__ Wqkv, const float* __restrict__ bqkv,
    const bf16* __restrict__ Xt,
    bf16* __restrict__ qg, bf16* __restrict__ kg, bf16* __restrict__ vh) {
  // double buffer: per buf At 128*32*2=8192B, Bt 64*32*2=4096B; Ot reuse 20480B
  __shared__ __align__(16) char sm[2 * 12288];
  const int tid = threadIdx.x;
  const int wv = tid >> 5, lane = tid & 31, g = lane >> 4, nl = lane & 15;
  const int b = blockIdx.z;
  const int p0 = blockIdx.x * 64;
  const int o0 = blockIdx.y * 128;
  const int wm = wv >> 1;   // 0..3
  const int wn = wv & 1;    // 0..1

  auto stage = [&](int kk, int bufi) {
    bf16* At = (bf16*)(sm + bufi * 12288);
    bf16* Bt = (bf16*)(sm + bufi * 12288 + 8192);
#pragma unroll
    for (int i = 0; i < 2; ++i) {
      int e = tid + i * 256;
      int row = e >> 2, seg = e & 3;
      async_ld_b128(At + row * 32 + seg * 8,
                    Wqkv + (size_t)(o0 + row) * CC + kk + seg * 8);
    }
    {
      int row = tid >> 2, seg = tid & 3;
      async_ld_b128(Bt + row * 32 + seg * 8,
                    Xt + ((size_t)b * NP + p0 + row) * CC + kk + seg * 8);
    }
  };

  v8f acc00 = {0,0,0,0,0,0,0,0}, acc01 = {0,0,0,0,0,0,0,0};
  v8f acc10 = {0,0,0,0,0,0,0,0}, acc11 = {0,0,0,0,0,0,0,0};

  stage(0, 0);
  wait_async0();
  __syncthreads();

  for (int kk = 0; kk < CC; kk += 32) {
    const int cur = (kk >> 5) & 1;
    if (kk + 32 < CC) stage(kk + 32, cur ^ 1);  // prefetch next k-step
    const bf16* At = (const bf16*)(sm + cur * 12288);
    const bf16* Bt = (const bf16*)(sm + cur * 12288 + 8192);
    v16bf a0 = frag_a(At + (wm * 32 + 0) * 32, 32);
    v16bf a1 = frag_a(At + (wm * 32 + 16) * 32, 32);
    v16bf b0 = frag_b(Bt + (wn * 32 + 0) * 32, 32);
    v16bf b1 = frag_b(Bt + (wn * 32 + 16) * 32, 32);
    acc00 = wmma_bf16(a0, b0, acc00);
    acc01 = wmma_bf16(a0, b1, acc01);
    acc10 = wmma_bf16(a1, b0, acc10);
    acc11 = wmma_bf16(a1, b1, acc11);
    wait_async0();
    __syncthreads();
  }

  if (o0 == 0) {
    // q (o<64) and k (64<=o<128): direct 16B c-contiguous stores
#pragma unroll
    for (int mi = 0; mi < 2; ++mi) {
#pragma unroll
      for (int ni = 0; ni < 2; ++ni) {
        const v8f& acc = mi ? (ni ? acc11 : acc10) : (ni ? acc01 : acc00);
        int ogl = (wm * 2 + mi) * 16 + g * 8;
        bf16* dstb = (ogl < 64) ? qg : kg;
        int oc = ogl & 63;
        int p = p0 + (wn * 2 + ni) * 16 + nl;
        union { bf16 h[8]; uint4 u; } pk;
#pragma unroll
        for (int r = 0; r < 8; ++r)
          pk.h[r] = (bf16)(acc[r] + bqkv[ogl + r]);
        *(uint4*)(dstb + ((size_t)b * NP + p) * CQ + oc) = pk.u;
      }
    }
  } else {
    // v: stage [o_local][p_local] in LDS, then async store-from-LDS rows
    bf16* Ot = (bf16*)sm;  // 128*80*2 = 20480B
#pragma unroll
    for (int mi = 0; mi < 2; ++mi) {
#pragma unroll
      for (int ni = 0; ni < 2; ++ni) {
        const v8f& acc = mi ? (ni ? acc11 : acc10) : (ni ? acc01 : acc00);
        int ol = (wm * 2 + mi) * 16 + g * 8;
        int pl = (wn * 2 + ni) * 16 + nl;
#pragma unroll
        for (int r = 0; r < 8; ++r)
          Ot[(ol + r) * 80 + pl] = (bf16)(acc[r] + bqkv[o0 + ol + r]);
      }
    }
    __syncthreads();
    const int cbase = o0 - 128;
    for (int ci = tid; ci < 1024; ci += 256) {
      int ol = ci >> 3, pseg = ci & 7;
      int p = p0 + pseg * 8;
      int h = p / WW, w = p % WW;
      async_st_b128(vh + (((size_t)b * CC + cbase + ol) * HH + h) * WW + w,
                    Ot + ol * 80 + pseg * 8);
    }
    // s_endpgm performs implicit wait-idle (covers ASYNCcnt)
  }
}

// ---------------------------------------------------------------------------
// K1b: vh [b][c][h][w] -> vw [b][c][w][h]  (async-load plane, LDS transpose)
// ---------------------------------------------------------------------------
__global__ __launch_bounds__(256) void transpose_v_kernel(
    const bf16* __restrict__ vh, bf16* __restrict__ vw) {
  __shared__ __align__(16) bf16 tile[96 * 104];
  const int tid = threadIdx.x;
  const size_t plane = blockIdx.x;
  const bf16* src = vh + plane * NP;
  bf16* dst = vw + plane * NP;
  for (int idx = tid; idx < 96 * 12; idx += 256) {
    int row = idx / 12, seg = idx % 12;
    async_ld_b128(tile + row * 104 + seg * 8, src + row * WW + seg * 8);
  }
  wait_async0();
  __syncthreads();
  for (int idx = tid; idx < WW * 48; idx += 256) {
    int w = idx / 48, hp = (idx % 48) * 2;
    union { bf16 h[2]; unsigned u; } pk;
    pk.h[0] = tile[hp * 104 + w];
    pk.h[1] = tile[(hp + 1) * 104 + w];
    *(unsigned*)(dst + w * HH + hp) = pk.u;
  }
}

// ---------------------------------------------------------------------------
// K2 pass1: per (b,w) column. eH (diag-masked) via WMMA, per-row (m,s) stats,
// unnormalized oH_un[c,h] via WMMA with double-buffered async V chunks.
// oH layout [b][h][w][c] f32. grid: B*W.
// ---------------------------------------------------------------------------
__global__ __launch_bounds__(256) void pass1_kernel(
    const bf16* __restrict__ qg, const bf16* __restrict__ kg,
    const bf16* __restrict__ vw,
    float* __restrict__ mH, float* __restrict__ sH, float* __restrict__ oH) {
  __shared__ __align__(16) bf16 Qt[96 * 64];
  __shared__ __align__(16) bf16 Kt[96 * 64];
  __shared__ __align__(16) bf16 P[96 * 112];
  __shared__ __align__(16) char scratch[96 * 100 * 4];  // E f32 | Vt x2 (14336 each)
  float* E = (float*)scratch;
  const int tid = threadIdx.x, wv = tid >> 5, lane = tid & 31;
  const int g = lane >> 4, nl = lane & 15;
  const int b = blockIdx.x / WW, w = blockIdx.x % WW;

  for (int idx = tid; idx < 96 * 8; idx += 256) {
    int row = idx >> 3, seg = idx & 7;
    size_t src = ((size_t)b * NP + row * WW + w) * CQ + seg * 8;
    async_ld_b128(Qt + row * CQ + seg * 8, qg + src);
    async_ld_b128(Kt + row * CQ + seg * 8, kg + src);
  }
  wait_async0();
  __syncthreads();

  for (int t = wv; t < 36; t += 8) {
    int tm = t / 6, tn = t % 6;
    v8f acc = {0, 0, 0, 0, 0, 0, 0, 0};
#pragma unroll
    for (int k0 = 0; k0 < 64; k0 += 32) {
      v16bf a = frag_a(Qt + tm * 16 * CQ + k0, CQ);
      v16bf bb = frag_b(Kt + tn * 16 * CQ + k0, CQ);
      acc = wmma_bf16(a, bb, acc);
    }
#pragma unroll
    for (int r = 0; r < 8; ++r) {
      int h = tm * 16 + g * 8 + r;
      int j = tn * 16 + nl;
      float vv = acc[r];
      if (h == j) vv = -1e30f;  // diagonal mask
      E[h * 100 + j] = vv;
    }
  }
  __syncthreads();

  if (tid < 96) {
    int h = tid;
    float m = -3.0e38f;
    for (int j = 0; j < 96; ++j) m = fmaxf(m, E[h * 100 + j]);
    float s = 0.f;
    for (int j = 0; j < 96; ++j) {
      float e = __expf(E[h * 100 + j] - m);
      s += e;
      P[h * 112 + j] = (bf16)e;
    }
    mH[((size_t)b * HH + h) * WW + w] = m;
    sH[((size_t)b * HH + h) * WW + w] = s;
  }
  __syncthreads();

  auto stage_v = [&](int c0, bf16* Vt) {
    for (int idx = tid; idx < 64 * 12; idx += 256) {
      int row = idx / 12, seg = idx % 12;
      async_ld_b128(Vt + row * 112 + seg * 8,
                    vw + (((size_t)b * CC + c0 + row) * WW + w) * HH + seg * 8);
    }
  };
  stage_v(0, (bf16*)scratch);
  wait_async0();
  __syncthreads();

  for (int c0 = 0; c0 < CC; c0 += 64) {
    const int cur = (c0 >> 6) & 1;
    const bf16* Vt = (const bf16*)(scratch + cur * 14336);
    if (c0 + 64 < CC) stage_v(c0 + 64, (bf16*)(scratch + (cur ^ 1) * 14336));
#pragma unroll
    for (int tt = 0; tt < 3; ++tt) {
      int t = wv * 3 + tt;
      int tm = t / 6, tn = t % 6;
      v8f acc = {0, 0, 0, 0, 0, 0, 0, 0};
#pragma unroll
      for (int j0 = 0; j0 < 96; j0 += 32) {
        v16bf a = frag_a(Vt + tm * 16 * 112 + j0, 112);
        v16bf bb = frag_b(P + tn * 16 * 112 + j0, 112);
        acc = wmma_bf16(a, bb, acc);
      }
      int c = c0 + tm * 16 + g * 8;
      int h = tn * 16 + nl;
      float* dst = oH + (((size_t)b * HH + h) * WW + w) * CC + c;
      *(float4*)dst = make_float4(acc[0], acc[1], acc[2], acc[3]);
      *(float4*)(dst + 4) = make_float4(acc[4], acc[5], acc[6], acc[7]);
    }
    wait_async0();
    __syncthreads();
  }
}

// ---------------------------------------------------------------------------
// K3 pass2: per (b,h) row. eW via WMMA, flash-style merge with pass1 stats,
// out = gamma*(oH_un*coefH + V@P_W^T) + x.  grid: B*H.
// ---------------------------------------------------------------------------
__global__ __launch_bounds__(256) void pass2_kernel(
    const bf16* __restrict__ qg, const bf16* __restrict__ kg,
    const bf16* __restrict__ vh,
    const float* __restrict__ mH, const float* __restrict__ sH,
    const float* __restrict__ oH, const float* __restrict__ x,
    const float* __restrict__ gamma, float* __restrict__ out) {
  __shared__ __align__(16) bf16 Qt[96 * 64];
  __shared__ __align__(16) bf16 Kt[96 * 64];
  __shared__ __align__(16) bf16 P[96 * 112];
  // E f32 (38400) overlaps [Vt0|Vt1 (14336 each) | obuf f32 (24832)]
  __shared__ __align__(16) char scratch[53504];
  __shared__ float coefH[96];
  float* E = (float*)scratch;
  float* obuf = (float*)(scratch + 28672);  // [64][97]
  const int tid = threadIdx.x, wv = tid >> 5, lane = tid & 31;
  const int g = lane >> 4, nl = lane & 15;
  const int b = blockIdx.x / HH, h = blockIdx.x % HH;

  for (int idx = tid; idx < 96 * 8; idx += 256) {
    int row = idx >> 3, seg = idx & 7;
    size_t src = ((size_t)b * NP + h * WW + row) * CQ + seg * 8;
    async_ld_b128(Qt + row * CQ + seg * 8, qg + src);
    async_ld_b128(Kt + row * CQ + seg * 8, kg + src);
  }
  wait_async0();
  __syncthreads();

  for (int t = wv; t < 36; t += 8) {
    int tm = t / 6, tn = t % 6;
    v8f acc = {0, 0, 0, 0, 0, 0, 0, 0};
#pragma unroll
    for (int k0 = 0; k0 < 64; k0 += 32) {
      v16bf a = frag_a(Qt + tm * 16 * CQ + k0, CQ);
      v16bf bb = frag_b(Kt + tn * 16 * CQ + k0, CQ);
      acc = wmma_bf16(a, bb, acc);
    }
#pragma unroll
    for (int r = 0; r < 8; ++r)
      E[(tm * 16 + g * 8 + r) * 100 + tn * 16 + nl] = acc[r];
  }
  __syncthreads();

  if (tid < 96) {
    int w = tid;
    float mW = -3.0e38f;
    for (int j = 0; j < 96; ++j) mW = fmaxf(mW, E[w * 100 + j]);
    float sW = 0.f;
    for (int j = 0; j < 96; ++j) sW += __expf(E[w * 100 + j] - mW);
    float mh = mH[((size_t)b * HH + h) * WW + w];
    float sh = sH[((size_t)b * HH + h) * WW + w];
    float M = fmaxf(mW, mh);
    float aW = __expf(mW - M), aH = __expf(mh - M);
    float denom = sh * aH + sW * aW;
    float invW = aW / denom;
    for (int j = 0; j < 96; ++j)
      P[w * 112 + j] = (bf16)(__expf(E[w * 100 + j] - mW) * invW);
    coefH[w] = aH / denom;
  }
  __syncthreads();

  auto stage_v = [&](int c0, bf16* Vt) {
    for (int idx = tid; idx < 64 * 12; idx += 256) {
      int row = idx / 12, seg = idx % 12;
      async_ld_b128(Vt + row * 112 + seg * 8,
                    vh + (((size_t)b * CC + c0 + row) * HH + h) * WW + seg * 8);
    }
  };
  stage_v(0, (bf16*)scratch);
  wait_async0();
  __syncthreads();

  const float gm = gamma[0];
  for (int c0 = 0; c0 < CC; c0 += 64) {
    const int cur = (c0 >> 6) & 1;
    const bf16* Vt = (const bf16*)(scratch + cur * 14336);
    if (c0 + 64 < CC) stage_v(c0 + 64, (bf16*)(scratch + (cur ^ 1) * 14336));
#pragma unroll
    for (int tt = 0; tt < 3; ++tt) {
      int t = wv * 3 + tt;
      int tm = t / 6, tn = t % 6;
      v8f acc = {0, 0, 0, 0, 0, 0, 0, 0};
#pragma unroll
      for (int j0 = 0; j0 < 96; j0 += 32) {
        v16bf a = frag_a(Vt + tm * 16 * 112 + j0, 112);
        v16bf bb = frag_b(P + tn * 16 * 112 + j0, 112);
        acc = wmma_bf16(a, bb, acc);
      }
      int c = c0 + tm * 16 + g * 8;
      int w = tn * 16 + nl;
      const float* ohp = oH + (((size_t)b * HH + h) * WW + w) * CC + c;
      float4 o1 = *(const float4*)ohp;
      float4 o2 = *(const float4*)(ohp + 4);
      float oh[8] = {o1.x, o1.y, o1.z, o1.w, o2.x, o2.y, o2.z, o2.w};
      float ch = coefH[w];
#pragma unroll
      for (int r = 0; r < 8; ++r)
        obuf[(tm * 16 + g * 8 + r) * 97 + w] = gm * (oh[r] * ch + acc[r]);
    }
    __syncthreads();
    for (int idx = tid; idx < 64 * 96; idx += 256) {
      int cl = idx / 96, w = idx % 96;
      size_t xoff = (((size_t)b * CC + c0 + cl) * HH + h) * WW + w;
      out[xoff] = obuf[cl * 97 + w] + x[xoff];
    }
    wait_async0();
    __syncthreads();
  }
}

// ---------------------------------------------------------------------------
extern "C" void kernel_launch(void* const* d_in, const int* in_sizes, int n_in,
                              void* d_out, int out_size, void* d_ws,
                              size_t ws_size, hipStream_t stream) {
  (void)in_sizes; (void)n_in; (void)out_size; (void)ws_size;
  const float* x     = (const float*)d_in[0];
  const float* Wq    = (const float*)d_in[1];
  const float* bq    = (const float*)d_in[2];
  const float* Wk    = (const float*)d_in[3];
  const float* bk    = (const float*)d_in[4];
  const float* Wv    = (const float*)d_in[5];
  const float* bv    = (const float*)d_in[6];
  const float* gamma = (const float*)d_in[7];
  float* out = (float*)d_out;

  char* ws = (char*)d_ws;
  size_t off = 0;
  auto carve = [&](size_t bytes) -> char* {
    char* p = ws + off;
    off += (bytes + 255) & ~(size_t)255;
    return p;
  };
  bf16*  Wqkv = (bf16*)carve((size_t)OC * CC * 2);
  float* bqkv = (float*)carve((size_t)OC * 4);
  bf16*  Xt   = (bf16*)carve((size_t)BB * NP * CC * 2);
  bf16*  qg   = (bf16*)carve((size_t)BB * NP * CQ * 2);
  bf16*  kg   = (bf16*)carve((size_t)BB * NP * CQ * 2);
  bf16*  vh   = (bf16*)carve((size_t)BB * CC * NP * 2);
  bf16*  vw   = (bf16*)carve((size_t)BB * CC * NP * 2);
  float* mHb  = (float*)carve((size_t)BB * NP * 4);
  float* sHb  = (float*)carve((size_t)BB * NP * 4);
  float* oHb  = (float*)carve((size_t)BB * NP * CC * 4);

  pack_weights_kernel<<<(OC * CC + 255) / 256, 256, 0, stream>>>(
      Wq, bq, Wk, bk, Wv, bv, Wqkv, bqkv);
  transpose_x_kernel<<<dim3(NP / 32, CC / 32, BB), 256, 0, stream>>>(x, Xt);
  qkv_gemm_kernel<<<dim3(NP / 64, OC / 128, BB), 256, 0, stream>>>(
      Wqkv, bqkv, Xt, qg, kg, vh);
  transpose_v_kernel<<<BB * CC, 256, 0, stream>>>(vh, vw);
  pass1_kernel<<<BB * WW, 256, 0, stream>>>(qg, kg, vw, mHb, sHb, oHb);
  pass2_kernel<<<BB * HH, 256, 0, stream>>>(qg, kg, vh, mHb, sHb, oHb, x,
                                            gamma, out);
}
